// SelfAttention_63926293234062
// MI455X (gfx1250) — matrix operand order repaired and checked by reading, not compiled, not run on hardware
//
#include <hip/hip_runtime.h>

// ---------------------------------------------------------------------------
// Self-attention (B=8, T=1024, D=1024, H=16, HD=64) for gfx1250 / MI455X.
// f32 -> bf16 WMMA (v_wmma_f32_16x16x32_bf16), f32 accumulation.
// [QKV GEMM 128x256 tiles] -> [flash attention, 64-key tiles] -> [proj + bias]
// CDNA5 async global->LDS copies (ASYNCcnt) with plain-load fallback.
// Workspace: 48MB bf16 QKV + 16MB bf16 attn-out = 64MB of d_ws.
// ---------------------------------------------------------------------------

typedef __bf16 v16bf __attribute__((ext_vector_type(16)));
typedef float  v8f   __attribute__((ext_vector_type(8)));
typedef int    v4i   __attribute__((vector_size(16)));

constexpr int kB  = 8;
constexpr int kT  = 1024;
constexpr int kD  = 1024;
constexpr int kH  = 16;
constexpr int kHD = 64;
constexpr int kM  = kB * kT;       // 8192
constexpr int kNqkv = 3 * kD;      // 3072
constexpr float kScale = 0.125f;   // 64^-0.5

// ---- CDNA5 async global->LDS (ASYNCcnt) availability probe ----------------
#if defined(__has_builtin)
#if __has_builtin(__builtin_amdgcn_global_load_async_to_lds_b128) && \
    __has_builtin(__builtin_amdgcn_s_wait_asynccnt)
#define USE_ASYNC_LDS 1
#endif
#endif
#ifndef USE_ASYNC_LDS
#define USE_ASYNC_LDS 0
#endif

#if USE_ASYNC_LDS
typedef __attribute__((address_space(1))) v4i GV4;  // global int4
typedef __attribute__((address_space(3))) v4i LV4;  // LDS int4
__device__ __forceinline__ void async_cp16(const void* g, void* l) {
  __builtin_amdgcn_global_load_async_to_lds_b128((GV4*)g, (LV4*)l, 0, 0);
}
__device__ __forceinline__ void async_drain() {
  __builtin_amdgcn_s_wait_asynccnt(0);
}
#endif

// A-matrix (16x32 bf16) per-lane element -> K index (contiguous, matches B):
//   K = 16*half + e
__device__ __forceinline__ int kidxA(int e, int hf) {
  return (hf << 4) + e;
}

// ---------------------------------------------------------------------------
// GEMM: Out[M,N] = A[M,K]*B[K,N] (+bias). Block 256 thr (8 waves),
// tile 128x256, BK=32. Wave tile 64x64 => 4x4 WMMA per K-step.
// B tile stored transposed+padded so B fragments are contiguous b128 reads.
// ---------------------------------------------------------------------------
template <bool A_BF16, bool OUT_BF16>
__global__ __launch_bounds__(256) void gemm_wmma(
    const void* __restrict__ Aop, const float* __restrict__ Bmat,
    const float* __restrict__ bias, void* __restrict__ Out,
    int M, int N, int K)
{
  constexpr int BM = 128, BN = 256, BK = 32, BKP = BK + 8;  // 80B row stride
  __shared__ __bf16 As[BM][BK];      // 8 KB
  __shared__ __bf16 Bst[BN][BKP];    // 20 KB, transposed [n][k]

  const int tid  = threadIdx.x;
  const int lane = tid & 31;
  const int wave = tid >> 5;
  const int ln   = lane & 15;
  const int hf   = lane >> 4;
  const int waveM = (wave >> 2) * 64;   // 0 or 64
  const int waveN = (wave & 3) * 64;    // 0,64,128,192

  const int m0 = blockIdx.x * BM;
  const int n0 = blockIdx.y * BN;

  v8f acc[4][4] = {};

  const int arow = tid >> 1;            // 0..127
  const int ak0  = (tid & 1) << 4;      // 0 or 16

  for (int k0 = 0; k0 < K; k0 += BK) {
    __syncthreads();
    // ---- A tile: 128x32, 16 contiguous bf16 per thread ----
    if constexpr (A_BF16) {
      const __bf16* ga = (const __bf16*)Aop + (size_t)(m0 + arow) * K + k0 + ak0;
#if USE_ASYNC_LDS
      async_cp16(ga,     &As[arow][ak0]);
      async_cp16(ga + 8, &As[arow][ak0 + 8]);
#else
#pragma unroll
      for (int i = 0; i < 16; ++i) As[arow][ak0 + i] = ga[i];
#endif
    } else {
      const float* ga = (const float*)Aop + (size_t)(m0 + arow) * K + k0 + ak0;
#pragma unroll
      for (int i = 0; i < 16; ++i) As[arow][ak0 + i] = (__bf16)ga[i];
    }
    // ---- B tile: one n-column per thread, coalesced across lanes ----
#pragma unroll
    for (int k = 0; k < BK; ++k)
      Bst[tid][k] = (__bf16)Bmat[(size_t)(k0 + k) * N + n0 + tid];
    if (k0 + BK < K) {  // global_prefetch_b8 of next tiles
      __builtin_prefetch(&Bmat[(size_t)(k0 + BK) * N + n0 + tid], 0, 1);
    }
#if USE_ASYNC_LDS
    if constexpr (A_BF16) async_drain();
#endif
    __syncthreads();

    // ---- fragments (all contiguous 16B runs in LDS) ----
    v16bf afr[4];
#pragma unroll
    for (int mt = 0; mt < 4; ++mt) {
      const int row = waveM + mt * 16 + ln;
#pragma unroll
      for (int e = 0; e < 16; ++e) afr[mt][e] = As[row][kidxA(e, hf)];
    }
    v16bf bfr[4];
#pragma unroll
    for (int nt = 0; nt < 4; ++nt) {
      const int col = waveN + nt * 16 + ln;
#pragma unroll
      for (int e = 0; e < 16; ++e) bfr[nt][e] = Bst[col][(hf << 4) + e];
    }
#pragma unroll
    for (int mt = 0; mt < 4; ++mt)
#pragma unroll
      for (int nt = 0; nt < 4; ++nt)
        acc[mt][nt] = __builtin_amdgcn_wmma_f32_16x16x32_bf16(
            false, afr[mt], false, bfr[nt], (short)0, acc[mt][nt], false, false);
  }

  // ---- epilogue (C layout: n = lane&15, m = r + 8*(lane>>4)) ----
#pragma unroll
  for (int mt = 0; mt < 4; ++mt) {
#pragma unroll
    for (int nt = 0; nt < 4; ++nt) {
      const int col = n0 + waveN + nt * 16 + ln;
#pragma unroll
      for (int r = 0; r < 8; ++r) {
        const int row = m0 + waveM + mt * 16 + r + (hf << 3);
        float v = acc[mt][nt][r];
        if constexpr (OUT_BF16) {
          ((__bf16*)Out)[(size_t)row * N + col] = (__bf16)v;
        } else {
          ((float*)Out)[(size_t)row * N + col] = v + bias[col];
        }
      }
    }
  }
}

// ---------------------------------------------------------------------------
// Flash attention over QKV (bf16, layout [B*T, 3*D]; head h: col s*D + h*64).
// Grid: (B*H, T/64). Block 128 thr = 4 waves; wave owns 16 q-rows.
// 64 keys per iteration: 8 score WMMAs + 8 PV WMMAs.
// ---------------------------------------------------------------------------
__global__ __launch_bounds__(128) void attn_flash(
    const __bf16* __restrict__ qkv, __bf16* __restrict__ attn_out)
{
  constexpr int KT = 64;             // keys per iteration
  __shared__ __bf16 Ks[KT][kHD];     // 8 KB
  __shared__ __bf16 Vs[KT][kHD];     // 8 KB
  __shared__ __bf16 Ps[4][16][KT];   // 8 KB  (per-wave P re-layout)

  const int bh = blockIdx.x;
  const int b  = bh / kH;
  const int h  = bh % kH;
  const int tid  = threadIdx.x;
  const int lane = tid & 31;
  const int wave = tid >> 5;
  const int ln   = lane & 15;
  const int hf   = lane >> 4;
  const int q0   = blockIdx.y * 64 + wave * 16;

  // Q fragments: A-layout, row = ln, two K=32 chunks over HD=64.
  v16bf qf[2];
  {
    const size_t qbase = ((size_t)(b * kT + q0 + ln)) * (3 * kD) + h * kHD;
#pragma unroll
    for (int c = 0; c < 2; ++c)
#pragma unroll
      for (int e = 0; e < 16; ++e)
        qf[c][e] = qkv[qbase + c * 32 + kidxA(e, hf)];
  }

  v8f o[4] = {};
  float mrow[8], lrow[8];
#pragma unroll
  for (int r = 0; r < 8; ++r) { mrow[r] = -1e30f; lrow[r] = 0.f; }

  const int krow = tid >> 1;         // 0..63
  const int kc0  = (tid & 1) << 5;   // 0 or 32 (32 contiguous bf16)

  for (int kt = 0; kt < kT / KT; ++kt) {
    __syncthreads();
    // ---- stage 64 keys/values of this head into LDS ----
    {
      const size_t gr =
          ((size_t)(b * kT + kt * KT + krow)) * (3 * kD) + h * kHD + kc0;
#if USE_ASYNC_LDS
      async_cp16(&qkv[gr + kD],           &Ks[krow][kc0]);
      async_cp16(&qkv[gr + kD + 8],       &Ks[krow][kc0 + 8]);
      async_cp16(&qkv[gr + kD + 16],      &Ks[krow][kc0 + 16]);
      async_cp16(&qkv[gr + kD + 24],      &Ks[krow][kc0 + 24]);
      async_cp16(&qkv[gr + 2 * kD],       &Vs[krow][kc0]);
      async_cp16(&qkv[gr + 2 * kD + 8],   &Vs[krow][kc0 + 8]);
      async_cp16(&qkv[gr + 2 * kD + 16],  &Vs[krow][kc0 + 16]);
      async_cp16(&qkv[gr + 2 * kD + 24],  &Vs[krow][kc0 + 24]);
      async_drain();
#else
#pragma unroll
      for (int i = 0; i < 32; ++i) {
        Ks[krow][kc0 + i] = qkv[gr + kD + i];
        Vs[krow][kc0 + i] = qkv[gr + 2 * kD + i];
      }
#endif
    }
    __syncthreads();

    // ---- S = Q*K^T : 4 key-tiles of 16, each over HD=64 (2 k-steps) ----
    v8f s[4] = {};
#pragma unroll
    for (int nt = 0; nt < 4; ++nt) {
      const int key = nt * 16 + ln;
#pragma unroll
      for (int c = 0; c < 2; ++c) {
        v16bf kf;  // B-layout: k = c*32 + 16*half + e (contiguous in LDS)
#pragma unroll
        for (int e = 0; e < 16; ++e)
          kf[e] = Ks[key][c * 32 + (hf << 4) + e];
        s[nt] = __builtin_amdgcn_wmma_f32_16x16x32_bf16(
            false, qf[c], false, kf, (short)0, s[nt], false, false);
      }
#pragma unroll
      for (int r = 0; r < 8; ++r) s[nt][r] *= kScale;
    }

    // ---- online softmax (row stats via 16-lane shuffle reduction) ----
    float mnew[8], alpha[8];
#pragma unroll
    for (int r = 0; r < 8; ++r) {
      float v = fmaxf(fmaxf(s[0][r], s[1][r]), fmaxf(s[2][r], s[3][r]));
      v = fmaxf(v, __shfl_xor(v, 1, 32));
      v = fmaxf(v, __shfl_xor(v, 2, 32));
      v = fmaxf(v, __shfl_xor(v, 4, 32));
      v = fmaxf(v, __shfl_xor(v, 8, 32));
      mnew[r]  = fmaxf(mrow[r], v);
      alpha[r] = __expf(mrow[r] - mnew[r]);
      mrow[r]  = mnew[r];
    }
#pragma unroll
    for (int nt = 0; nt < 4; ++nt)
#pragma unroll
      for (int r = 0; r < 8; ++r) s[nt][r] = __expf(s[nt][r] - mnew[r]);
#pragma unroll
    for (int r = 0; r < 8; ++r) {
      float v = (s[0][r] + s[1][r]) + (s[2][r] + s[3][r]);
      v += __shfl_xor(v, 1, 32);
      v += __shfl_xor(v, 2, 32);
      v += __shfl_xor(v, 4, 32);
      v += __shfl_xor(v, 8, 32);
      lrow[r] = lrow[r] * alpha[r] + v;
    }
#pragma unroll
    for (int ot = 0; ot < 4; ++ot)
#pragma unroll
      for (int r = 0; r < 8; ++r) o[ot][r] *= alpha[r];

    // ---- re-layout P: C-format accum -> LDS -> A-format fragments ----
#pragma unroll
    for (int nt = 0; nt < 4; ++nt)
#pragma unroll
      for (int r = 0; r < 8; ++r)
        Ps[wave][r + (hf << 3)][nt * 16 + ln] = (__bf16)s[nt][r];
    __syncthreads();

    v16bf pf[2];
#pragma unroll
    for (int c = 0; c < 2; ++c)
#pragma unroll
      for (int e = 0; e < 16; ++e)
        pf[c][e] = Ps[wave][ln][c * 32 + kidxA(e, hf)];

    // ---- O += P*V : 4 hd-tiles x 2 key-chunks ----
#pragma unroll
    for (int ot = 0; ot < 4; ++ot) {
      const int hd = ot * 16 + ln;
#pragma unroll
      for (int c = 0; c < 2; ++c) {
        v16bf vf;
#pragma unroll
        for (int e = 0; e < 16; ++e)
          vf[e] = Vs[c * 32 + (hf << 4) + e][hd];
        o[ot] = __builtin_amdgcn_wmma_f32_16x16x32_bf16(
            false, pf[c], false, vf, (short)0, o[ot], false, false);
      }
    }
  }

  // ---- normalize, write bf16 attention output [B*T, D] ----
#pragma unroll
  for (int ot = 0; ot < 4; ++ot) {
#pragma unroll
    for (int r = 0; r < 8; ++r) {
      const int q   = q0 + r + (hf << 3);
      const int col = h * kHD + ot * 16 + ln;
      attn_out[(size_t)(b * kT + q) * kD + col] = (__bf16)(o[ot][r] / lrow[r]);
    }
  }
}

// ---------------------------------------------------------------------------
extern "C" void kernel_launch(void* const* d_in, const int* in_sizes, int n_in,
                              void* d_out, int out_size, void* d_ws, size_t ws_size,
                              hipStream_t stream) {
  const float* x      = (const float*)d_in[0];   // [8192, 1024]
  const float* w_qkv  = (const float*)d_in[1];   // [1024, 3072]
  const float* w_proj = (const float*)d_in[2];   // [1024, 1024]
  const float* b_proj = (const float*)d_in[3];   // [1024]
  float* out = (float*)d_out;                    // [8192, 1024]

  __bf16* qkv  = (__bf16*)d_ws;                  // 48 MB
  __bf16* attn = qkv + (size_t)kM * kNqkv;       // +16 MB

  // 1) QKV projection -> bf16 workspace.
  {
    dim3 grid(kM / 128, kNqkv / 256);   // 64 x 12
    gemm_wmma<false, true><<<grid, 256, 0, stream>>>(
        (const void*)x, w_qkv, nullptr, (void*)qkv, kM, kNqkv, kD);
  }
  // 2) Flash attention.
  {
    dim3 grid(kB * kH, kT / 64);        // 128 x 16
    attn_flash<<<grid, 128, 0, stream>>>(qkv, attn);
  }
  // 3) Output projection + bias -> f32.
  {
    dim3 grid(kM / 128, kD / 256);      // 64 x 4
    gemm_wmma<true, false><<<grid, 256, 0, stream>>>(
        (const void*)attn, w_proj, b_proj, (void*)out, kM, kD, kD);
  }
}
